// TokenSoftAlignLoss_25941602468413
// MI455X (gfx1250) — matrix operand order repaired and checked by reading, not compiled, not run on hardware
//
#include <hip/hip_runtime.h>

// ---------------------------------------------------------------------------
// TokenSoftAlignLoss fused kernel for gfx1250 (MI455X, wave32, WMMA).
//   B=8, T=2048, D=512, fp32 in, (loss, tas) fp32 out.
// Math identity: the softmax denominator cancels inside cosine similarity,
// so we only accumulate num[t,:] = sum_s valid_s * exp(5*S_ts) * E_s in one
// streaming pass (no online max needed: |S|<=1 after L2 normalization).
//
// v2 changes vs v1:
//   * M-tile per block 16 -> 32 rows: B-fragments (En / EnT streams) are
//     reused across two row-tiles, halving L2 read traffic (~4GB -> ~2GB).
//   * Ping-pong split-K buffer sS[2][..]: one barrier per 32-column step
//     instead of two.
//   * global_prefetch hints for the next iteration's B streams.
// ---------------------------------------------------------------------------

#define B_  8
#define T_  2048
#define D_  512
#define TAUF 5.0f

typedef __attribute__((ext_vector_type(16))) __bf16 v16bf;
typedef __attribute__((ext_vector_type(8)))  __bf16 v8bf;
typedef __attribute__((ext_vector_type(8)))  float  v8f;

__device__ __forceinline__ v16bf cat8(v8bf lo, v8bf hi) {
    return __builtin_shufflevector(lo, hi, 0,1,2,3,4,5,6,7,8,9,10,11,12,13,14,15);
}
// 16 contiguous bf16 (two 16B-aligned chunks) -> one fragment register set
__device__ __forceinline__ v16bf ld16(const __bf16* p) {
    v8bf lo = *(const v8bf*)p;
    v8bf hi = *(const v8bf*)(p + 8);
    return cat8(lo, hi);
}

// ---------------------------------------------------------------------------
// Kernel 0: zero the loss accumulator (must rerun every launch for replay).
// ---------------------------------------------------------------------------
__global__ void init_kernel(float* accum) { accum[0] = 0.0f; }

// ---------------------------------------------------------------------------
// Kernel 1: per-batch "any valid column" flag. mask: 1 byte/elem, True=PAD.
// ---------------------------------------------------------------------------
__global__ void mask_kernel(const unsigned char* __restrict__ mask,
                            int* __restrict__ batchValid) {
    __shared__ int flag;
    const int b = blockIdx.x;
    if (threadIdx.x == 0) flag = 0;
    __syncthreads();
    int local = 0;
    for (int t = threadIdx.x; t < T_; t += blockDim.x)
        if (mask[(size_t)b * T_ + t] == 0) local = 1;
    if (local) atomicOr(&flag, 1);
    __syncthreads();
    if (threadIdx.x == 0) batchValid[b] = flag;
}

// ---------------------------------------------------------------------------
// Kernel 2: L2-normalize rows of P and E (fp32), emit bf16 Pn, En and EnT.
// One row per wave, 16 contiguous elements per lane.
// ---------------------------------------------------------------------------
__global__ void norm_kernel(const float* __restrict__ P,
                            const float* __restrict__ E,
                            __bf16* __restrict__ Pn,
                            __bf16* __restrict__ En,
                            __bf16* __restrict__ EnT) {
    const int wave = threadIdx.x >> 5;
    const int lane = threadIdx.x & 31;
    const int row  = blockIdx.x * 4 + wave;           // grid sized exactly
    const bool isE = row >= B_ * T_;
    const size_t rr = isE ? (size_t)(row - B_ * T_) : (size_t)row;
    const float* src = (isE ? E : P) + rr * D_ + lane * 16;

    float x[16];
    const float4* sp = (const float4*)src;
    #pragma unroll
    for (int i = 0; i < 4; ++i) {
        float4 v = sp[i];
        x[4*i+0] = v.x; x[4*i+1] = v.y; x[4*i+2] = v.z; x[4*i+3] = v.w;
    }
    float s = 0.f;
    #pragma unroll
    for (int i = 0; i < 16; ++i) s += x[i] * x[i];
    #pragma unroll
    for (int off = 16; off > 0; off >>= 1) s += __shfl_xor(s, off, 32);
    const float inv = 1.0f / fmaxf(sqrtf(s), 1e-12f);

    __bf16 o[16];
    #pragma unroll
    for (int i = 0; i < 16; ++i) o[i] = (__bf16)(x[i] * inv);

    __bf16* dst = (isE ? En : Pn) + rr * D_ + lane * 16;
    *(v8bf*)dst       = *(v8bf*)&o[0];
    *(v8bf*)(dst + 8) = *(v8bf*)&o[8];

    if (isE) {  // scatter transpose: EnT[b][d][t] (L2 combines the 2B writes)
        const int b = (int)(rr / T_);
        const int t = (int)(rr % T_);
        #pragma unroll
        for (int i = 0; i < 16; ++i) {
            const int d = lane * 16 + i;
            EnT[((size_t)b * D_ + d) * T_ + t] = o[i];
        }
    }
}

// ---------------------------------------------------------------------------
// Kernel 3: fused attention + cosine loss.
// Block = (batch b, 32-row tile). 128 threads = 4 waves.
//   GEMM1 (S = Pn * En^T): split-K across waves (each wave K=128 of D=512),
//          2 row-tiles x 2 N-tiles per 32-column step, reduced via LDS
//          (ping-pong buffer, single barrier per step).
//   GEMM2 (num += W * En): each wave owns a 128-wide D chunk, 2 row-tiles
//          (16 persistent C tiles = 128 VGPRs), B fragments shared between
//          the two row-tiles.
// Fragment layouts per CDNA5 ISA 7.12.2 (wave32):
//   A 16x32 bf16: lane(half=L/16, m=L&15), elem e -> k = (e<8 ? half*8+e
//                                                        : 16+half*8+(e-8))
//   B 32x16 bf16: lane(half, n=L&15),      elem e -> k = same as A (e<8 ? half*8+e : 16+half*8+(e-8))
//   C 16x16 f32 : vgpr r, lane -> (m = r + 8*half, n = L&15)
// ---------------------------------------------------------------------------
__global__ __launch_bounds__(128)
void main_kernel(const __bf16* __restrict__ Pn,
                 const __bf16* __restrict__ En,
                 const __bf16* __restrict__ EnT,
                 const unsigned char* __restrict__ mask,
                 const int* __restrict__ batchValid,
                 float* __restrict__ accum) {
    __shared__ __align__(16) __bf16 sPn[32 * D_];          // 32 KB P tile (bf16)
    __shared__ __align__(16) float  sS[2][4][32][32];      // 32 KB split-K ping-pong
    __shared__ __align__(16) unsigned char sMask[T_];      // 2 KB pad-mask row
    __shared__ float sDot[32], sNsq[32];

    const int tid  = threadIdx.x;
    const int wv   = tid >> 5;
    const int lane = tid & 31;
    const int half = lane >> 4;
    const int lm   = lane & 15;          // m for A/C fragments, n for B fragments
    const int b    = blockIdx.x >> 6;    // T_/32 = 64 tiles per batch
    const int t0   = (blockIdx.x & 63) * 32;

    // ---- stage P tile (bf16) + mask row into LDS -------------------------
    {
        const int4* g = (const int4*)(Pn + ((size_t)b * T_ + t0) * D_);
        int4* l = (int4*)sPn;
        #pragma unroll
        for (int i = 0; i < 16; ++i) l[tid + i * 128] = g[tid + i * 128];
        ((int4*)sMask)[tid] = ((const int4*)(mask + (size_t)b * T_))[tid];
        if (tid < 32) { sDot[tid] = 0.f; sNsq[tid] = 0.f; }
    }
    __syncthreads();

    // persistent accumulators: num[rt*16 + m, d], d in [wv*128, wv*128+128)
    v8f acc[2][8];
    #pragma unroll
    for (int rt = 0; rt < 2; ++rt)
        #pragma unroll
        for (int nt = 0; nt < 8; ++nt)
            acc[rt][nt] = (v8f){0.f,0.f,0.f,0.f,0.f,0.f,0.f,0.f};

    const size_t rowBase = (size_t)b * T_;     // En row index base
    const size_t dBase   = (size_t)b * D_;     // EnT row index base

    for (int it = 0; it < T_ / 32; ++it) {
        const int s0  = it * 32;
        const int buf = it & 1;

        // prefetch next iteration's B streams into L2/WGP$
        if (s0 + 32 < T_) {
            __builtin_prefetch(En  + (rowBase + s0 + 32 + lane) * D_ + wv * 128, 0, 1);
            __builtin_prefetch(EnT + (dBase + wv * 128 + lane) * T_ + s0 + 32, 0, 1);
        }

        // -------- GEMM1: partial S (2 row-tiles x 32 cols), K = wave's 128 --
        v8f c00 = (v8f){0.f,0.f,0.f,0.f,0.f,0.f,0.f,0.f};
        v8f c01 = c00, c10 = c00, c11 = c00;
        #pragma unroll
        for (int kk = 0; kk < 4; ++kk) {
            const int d0 = wv * 128 + kk * 32;
            const __bf16* ap0 = sPn + lm * D_        + d0 + half * 8;
            const __bf16* ap1 = sPn + (16 + lm) * D_ + d0 + half * 8;
            v16bf a0 = cat8(*(const v8bf*)ap0, *(const v8bf*)(ap0 + 16));
            v16bf a1 = cat8(*(const v8bf*)ap1, *(const v8bf*)(ap1 + 16));
            const __bf16* bp0 = En + (rowBase + s0 +      lm) * D_ + d0 + half * 8;
            const __bf16* bp1 = En + (rowBase + s0 + 16 + lm) * D_ + d0 + half * 8;
            v16bf b0 = cat8(*(const v8bf*)bp0, *(const v8bf*)(bp0 + 16));
            v16bf b1 = cat8(*(const v8bf*)bp1, *(const v8bf*)(bp1 + 16));
            c00 = __builtin_amdgcn_wmma_f32_16x16x32_bf16(false, a0, false, b0,
                                                          (short)0, c00, false, false);
            c01 = __builtin_amdgcn_wmma_f32_16x16x32_bf16(false, a0, false, b1,
                                                          (short)0, c01, false, false);
            c10 = __builtin_amdgcn_wmma_f32_16x16x32_bf16(false, a1, false, b0,
                                                          (short)0, c10, false, false);
            c11 = __builtin_amdgcn_wmma_f32_16x16x32_bf16(false, a1, false, b1,
                                                          (short)0, c11, false, false);
        }
        // ping-pong buffer: stores race with nothing (see barrier proof below)
        #pragma unroll
        for (int r = 0; r < 8; ++r) {
            const int m = r + 8 * half;
            sS[buf][wv][m][lm]           = c00[r];
            sS[buf][wv][m][16 + lm]      = c01[r];
            sS[buf][wv][16 + m][lm]      = c10[r];
            sS[buf][wv][16 + m][16 + lm] = c11[r];
        }
        // Single barrier: orders this iteration's stores before its reads.
        // Next iteration stores the *other* buffer, and any wave reaching
        // those stores has passed this barrier, which post-dates all reads
        // of that buffer from the previous iteration.
        __syncthreads();

        // ------ reduce split-K, build exp-weight A fragments (all waves) ---
        const int k1 = half * 8;
        const int k2 = 16 + half * 8;
        v16bf af[2];
        #pragma unroll
        for (int rt = 0; rt < 2; ++rt) {
            const int row = rt * 16 + lm;
            float ch1[8], ch2[8];
            #pragma unroll
            for (int j = 0; j < 8; ++j) { ch1[j] = 0.f; ch2[j] = 0.f; }
            #pragma unroll
            for (int w2 = 0; w2 < 4; ++w2) {
                const float* rowp = &sS[buf][w2][row][0];
                #pragma unroll
                for (int j = 0; j < 8; ++j) {
                    ch1[j] += rowp[k1 + j];
                    ch2[j] += rowp[k2 + j];
                }
            }
            #pragma unroll
            for (int j = 0; j < 8; ++j) {
                const float w1 = sMask[s0 + k1 + j] ? 0.f : __expf(TAUF * ch1[j]);
                const float w2 = sMask[s0 + k2 + j] ? 0.f : __expf(TAUF * ch2[j]);
                af[rt][j]     = (__bf16)w1;
                af[rt][8 + j] = (__bf16)w2;
            }
        }

        // ------ GEMM2: num += W * En ; B fragment shared by both row-tiles --
        #pragma unroll
        for (int nt = 0; nt < 8; ++nt) {
            const int d = wv * 128 + nt * 16 + lm;                 // n = lm
            const __bf16* bq = EnT + (dBase + d) * T_ + s0 + half * 8;
            v16bf bb = cat8(*(const v8bf*)bq, *(const v8bf*)(bq + 16));
            acc[0][nt] = __builtin_amdgcn_wmma_f32_16x16x32_bf16(false, af[0], false, bb,
                                                                 (short)0, acc[0][nt],
                                                                 false, false);
            acc[1][nt] = __builtin_amdgcn_wmma_f32_16x16x32_bf16(false, af[1], false, bb,
                                                                 (short)0, acc[1][nt],
                                                                 false, false);
        }
    }

    // ---- cosine: dot(Pn, num) and |num|^2 per row -------------------------
    #pragma unroll
    for (int rt = 0; rt < 2; ++rt) {
        float dotp[8], nsqp[8];
        #pragma unroll
        for (int r = 0; r < 8; ++r) { dotp[r] = 0.f; nsqp[r] = 0.f; }
        #pragma unroll
        for (int nt = 0; nt < 8; ++nt) {
            const int d = wv * 128 + nt * 16 + lm;
            #pragma unroll
            for (int r = 0; r < 8; ++r) {
                const float v  = acc[rt][nt][r];
                const float pv = (float)sPn[(rt * 16 + r + 8 * half) * D_ + d];
                dotp[r] += v * pv;
                nsqp[r] += v * v;
            }
        }
        #pragma unroll
        for (int r = 0; r < 8; ++r) {
            atomicAdd(&sDot[rt * 16 + r + 8 * half], dotp[r]);
            atomicAdd(&sNsq[rt * 16 + r + 8 * half], nsqp[r]);
        }
    }
    __syncthreads();

    if (tid < 32) {
        // |P|=1 after normalization; softmax denominator cancels in cos.
        const float nn  = sqrtf(sNsq[tid]);
        const float cos = sDot[tid] / fmaxf(nn, 1e-8f);
        const float contrib = (1.0f - cos) * (batchValid[b] ? 1.0f : 0.0f);
        atomicAdd(accum, contrib);
    }
}

// ---------------------------------------------------------------------------
// Kernel 4: finalize (loss, tas).
// ---------------------------------------------------------------------------
__global__ void finalize_kernel(const float* __restrict__ accum,
                                const int* __restrict__ batchValid,
                                float* __restrict__ out) {
    int nb = 0;
    #pragma unroll
    for (int b = 0; b < B_; ++b) nb += batchValid[b] ? 1 : 0;
    const float cnt  = (float)T_ * (float)nb;
    const float loss = (cnt > 0.f) ? accum[0] / fmaxf(cnt, 1.0f) : 0.0f;
    out[0] = loss;
    out[1] = (cnt > 0.f) ? (1.0f - loss) : loss;
}

// ---------------------------------------------------------------------------
extern "C" void kernel_launch(void* const* d_in, const int* in_sizes, int n_in,
                              void* d_out, int out_size, void* d_ws, size_t ws_size,
                              hipStream_t stream) {
    const float* P = (const float*)d_in[0];
    const float* E = (const float*)d_in[1];
    const unsigned char* mask = (const unsigned char*)d_in[2];   // numpy bool = 1B

    const size_t nPE = (size_t)B_ * T_ * D_;
    __bf16* Pn  = (__bf16*)d_ws;          // 16 MB
    __bf16* En  = Pn + nPE;               // 16 MB
    __bf16* EnT = En + nPE;               // 16 MB
    float*  accum      = (float*)(EnT + nPE);
    int*    batchValid = (int*)(accum + 4);

    init_kernel<<<1, 1, 0, stream>>>(accum);
    mask_kernel<<<B_, 256, 0, stream>>>(mask, batchValid);
    norm_kernel<<<(2 * B_ * T_) / 4, 128, 0, stream>>>(P, E, Pn, En, EnT);
    main_kernel<<<B_ * (T_ / 32), 128, 0, stream>>>(Pn, En, EnT, mask,
                                                    batchValid, accum);
    finalize_kernel<<<1, 1, 0, stream>>>(accum, batchValid, (float*)d_out);
}
